// LSAQLinear_33509334843613
// MI455X (gfx1250) — matrix operand order repaired and checked by reading, not compiled, not run on hardware
//
#include <hip/hip_runtime.h>
#include <hip/hip_fp16.h>

typedef __attribute__((ext_vector_type(16))) _Float16 v16h;
typedef __attribute__((ext_vector_type(8)))  _Float16 v8h;
typedef __attribute__((ext_vector_type(2)))  _Float16 v2h;
typedef __attribute__((ext_vector_type(8)))  float    v8f;
typedef __attribute__((ext_vector_type(4)))  int      v4i;

// Problem dims (fixed by the reference)
constexpr int Mdim = 4 * 2048;   // B*S = 8192
constexpr int Ndim = 11008;      // OUT
constexpr int Kdim = 4096;       // IN

// Tiling
constexpr int BM = 128, BN = 128, BK = 32;
constexpr int NK = Kdim / BK;          // 128 k-steps
constexpr int LDSTR = BK + 8;          // padded LDS row stride (halves): 80B rows

// int4 pair -> fp16 pair via mantissa-splice + packed FMA.
//   nibble n encodes w = (n^8)-8 + ... : bitcast(0x6400 | (n^8)) = 1024 + (w+8)
//   result = t*s + (-1032*s)  ==  w*s      (one v_pk_fma_f16 per byte)
__device__ __forceinline__ v2h dq_pair(int b, v2h s2, v2h c2) {
  unsigned t = ((unsigned)b & 0xFu) | (((unsigned)b << 12) & 0xF0000u);
  t ^= 0x64086408u;                       // fold (|0x6400) and (^8) for both halves
  v2h ht = __builtin_bit_cast(v2h, t);
  return ht * s2 + c2;                    // contracts to v_pk_fma_f16
}

__device__ __forceinline__ v16h cat16(v8h lo, v8h hi) {
  return __builtin_shufflevector(lo, hi, 0, 1, 2, 3, 4, 5, 6, 7,
                                         8, 9, 10, 11, 12, 13, 14, 15);
}

__launch_bounds__(256, 2)
__global__ void int4_linear_wmma(const __half* __restrict__ xh,
                                 const int*    __restrict__ wq,
                                 const __half* __restrict__ sch,
                                 const __half* __restrict__ bh,
                                 __half*       __restrict__ yh) {
  const _Float16* x      = reinterpret_cast<const _Float16*>(xh);
  const _Float16* scales = reinterpret_cast<const _Float16*>(sch);
  const _Float16* bias   = reinterpret_cast<const _Float16*>(bh);
  _Float16*       y      = reinterpret_cast<_Float16*>(yh);

  __shared__ _Float16 As[2][BM * LDSTR];   // activations  (fp16), filled by async loads
  __shared__ _Float16 Bs[2][BN * LDSTR];   // dequantized weights (fp16)

  const int tid  = threadIdx.x;
  const int lane = tid & 31;
  const int wave = tid >> 5;        // 8 waves
  const int wm   = wave & 1;        // 2 waves tile M (64 rows each)
  const int wn   = wave >> 1;       // 4 waves tile N (32 cols each)
  const int lrow = lane & 15;
  const int lhi  = lane >> 4;

  const int blockN = blockIdx.x * BN;
  const int blockM = blockIdx.y * BM;

  // Staging assignment: 2 threads per tile row, each covers 16 K-values.
  const int r = tid >> 1;           // 0..127
  const int h = tid & 1;            // 0..1

  const _Float16* agp = x  + (size_t)(blockM + r) * Kdim + h * 16;
  const int*      wgp = wq + (size_t)(blockN + r) * (Kdim / 2) + h * 8;

  const _Float16 sH = scales[blockN + r];                 // per-output-channel scale
  const _Float16 cH = (_Float16)(-1032.0f * (float)sH);   // fused (-1032*s) term
  v2h s2; s2[0] = sH; s2[1] = sH;
  v2h c2; c2[0] = cH; c2[1] = cH;

  v4i p0, p1;

  // A tile: per-lane async copy global -> LDS (ASYNCcnt), no VGPR staging.
  auto async_a = [&](int kt, int buf) {
    unsigned ldsa = (unsigned)(uintptr_t)&As[buf][r * LDSTR + h * 16];
    unsigned long long ga =
        (unsigned long long)(uintptr_t)(agp + (size_t)kt * BK);
    asm volatile(
        "global_load_async_to_lds_b128 %0, %1, off\n\t"
        "global_load_async_to_lds_b128 %0, %1, off offset:16"
        :: "v"(ldsa), "v"(ga) : "memory");
  };

  auto load_b_regs = [&](int kt) {
    const int* wp = wgp + kt * (BK / 2);
    p0 = *(const v4i*)(wp);                 // global_load_b128 (8 packed bytes)
    p1 = *(const v4i*)(wp + 4);
    if (kt + 4 < NK) {                      // global_prefetch_b8 into GL2
      __builtin_prefetch(agp + (kt + 4) * BK, 0, 1);
      __builtin_prefetch(wgp + (kt + 4) * (BK / 2), 0, 1);
    }
  };

  auto store_b = [&](int buf) {
    v8h blo, bhi;
#pragma unroll
    for (int i = 0; i < 4; ++i) {
      v2h d = dq_pair(p0[i], s2, c2);
      blo[2 * i] = d[0]; blo[2 * i + 1] = d[1];
    }
#pragma unroll
    for (int i = 0; i < 4; ++i) {
      v2h d = dq_pair(p1[i], s2, c2);
      bhi[2 * i] = d[0]; bhi[2 * i + 1] = d[1];
    }
    _Float16* bs = &Bs[buf][r * LDSTR + h * 16];
    *(v8h*)(bs)     = blo;                  // ds_store_b128 x2
    *(v8h*)(bs + 8) = bhi;
  };

  v8f acc[4][2];
#pragma unroll
  for (int i = 0; i < 4; ++i)
#pragma unroll
    for (int j = 0; j < 2; ++j)
      acc[i][j] = (v8f)(0.0f);

  // Prologue: stage tile 0.
  async_a(0, 0);
  load_b_regs(0);
  store_b(0);
  asm volatile("s_wait_asynccnt 0x0" ::: "memory");  // A tile 0 resident in LDS

  for (int kt = 0; kt < NK; ++kt) {
    const int cur = kt & 1;
    __syncthreads();   // publish tile kt (ds stores + waited asyncs); retire readers of cur^1

    if (kt + 1 < NK) {
      async_a(kt + 1, cur ^ 1);   // overlaps with the 8 WMMAs below
      load_b_regs(kt + 1);
    }

    // --- A fragments: 16x32 fp16, documented per-lane layout ---
    v16h afr[4];
#pragma unroll
    for (int i = 0; i < 4; ++i) {
      const _Float16* p = &As[cur][(wm * 64 + i * 16 + lrow) * LDSTR];
      v8h lo = *(const v8h*)(p + lhi * 8);        // K {0..7} or {8..15}
      v8h hi = *(const v8h*)(p + 16 + lhi * 8);   // K {16..23} or {24..31}
      afr[i] = cat16(lo, hi);
    }
    // --- B fragments: lane = N, K in same per-lane layout as A ---
    v16h bfr[2];
#pragma unroll
    for (int j = 0; j < 2; ++j) {
      const _Float16* p =
          &Bs[cur][(wn * 32 + j * 16 + lrow) * LDSTR];
      v8h lo = *(const v8h*)(p + lhi * 8);
      v8h hi = *(const v8h*)(p + 16 + lhi * 8);
      bfr[j] = cat16(lo, hi);
    }

#pragma unroll
    for (int i = 0; i < 4; ++i)
#pragma unroll
      for (int j = 0; j < 2; ++j)
        acc[i][j] = __builtin_amdgcn_wmma_f32_16x16x32_f16(
            false, afr[i], false, bfr[j], (short)0, acc[i][j], false, false);

    if (kt + 1 < NK) store_b(cur ^ 1);
    asm volatile("s_wait_asynccnt 0x0" ::: "memory");  // next A tile landed
  }

  // Epilogue: C/D layout — VGPR v holds M = v + 8*(lane>=16), N = lane&15.
#pragma unroll
  for (int j = 0; j < 2; ++j) {
    const int gn   = blockN + wn * 32 + j * 16 + lrow;
    const float bf = (float)bias[gn];
#pragma unroll
    for (int i = 0; i < 4; ++i) {
      const int gm0 = blockM + wm * 64 + i * 16 + lhi * 8;
#pragma unroll
      for (int v = 0; v < 8; ++v) {
        y[(size_t)(gm0 + v) * Ndim + gn] = (_Float16)(acc[i][j][v] + bf);
      }
    }
  }
}

extern "C" void kernel_launch(void* const* d_in, const int* in_sizes, int n_in,
                              void* d_out, int out_size, void* d_ws, size_t ws_size,
                              hipStream_t stream) {
  (void)in_sizes; (void)n_in; (void)d_ws; (void)ws_size; (void)out_size;
  const __half* x      = (const __half*)d_in[0];   // [B,S,IN] fp16
  const int*    wq     = (const int*)d_in[1];      // [OUT, IN/2] packed int4 (one byte per int32)
  const __half* scales = (const __half*)d_in[2];   // [OUT,1] fp16
  const __half* bias   = (const __half*)d_in[3];   // [1,OUT] fp16
  __half*       yout   = (__half*)d_out;           // [B,S,OUT] fp16

  dim3 grid(Ndim / BN, Mdim / BM);   // (86, 64)
  dim3 block(256);
  int4_linear_wmma<<<grid, block, 0, stream>>>(x, wq, scales, bias, yout);
}
